// GATSimilarity_61864708932306
// MI455X (gfx1250) — compile-verified, matrix operand order checked by reading, not run on hardware
//
#include <hip/hip_runtime.h>
#include <hip/hip_bf16.h>

// ---------------------------------------------------------------------------
// GAT similarity for MI455X (gfx1250, wave32).  GEMMs via v_wmma_f32_16x16x32_f16,
// segment softmax + aggregation via L2 float atomics.
// ---------------------------------------------------------------------------

#define IN_DIM 256
#define HID    128
#define HEADS  3
#define D1     (HEADS * HID)   // 384
#define NEG_SLOPE 0.2f

typedef __attribute__((ext_vector_type(16))) _Float16 v16h;
typedef __attribute__((ext_vector_type(8)))  _Float16 v8h;
typedef __attribute__((ext_vector_type(8)))  float    v8f;

// ---------------- helpers ----------------

__device__ __forceinline__ void get_edge(const int* __restrict__ ei, int E, int Nn,
                                         int e, int& s, int& d) {
  if (e < E) { s = ei[e]; d = ei[E + e]; }
  else       { s = e - E; d = e - E; }       // appended self loops
}

// correct concurrent float max via sign-split int atomics
__device__ __forceinline__ void atomicMaxFloat(float* addr, float v) {
  if (v >= 0.0f) atomicMax((int*)addr, __float_as_int(v));
  else           atomicMin((unsigned int*)addr, (unsigned int)__float_as_int(v));
}

__device__ __forceinline__ float leaky(float v) {
  return v > 0.0f ? v : NEG_SLOPE * v;
}

// ---------------- conversion / init ----------------

__global__ void fill_f32_kernel(float* __restrict__ p, int n, float v) {
  for (int i = blockIdx.x * blockDim.x + threadIdx.x; i < n; i += gridDim.x * blockDim.x)
    p[i] = v;
}

__global__ void cvt_x_kernel(const float* __restrict__ x, _Float16* __restrict__ xh, int n) {
  int i = blockIdx.x * blockDim.x + threadIdx.x;
  if (i < n) xh[i] = (_Float16)x[i];
}

// W1[256,384] -> W1t[384][256] f16 ; W2[384,128] -> W2t[128][384] f16  (transposed)
__global__ void cvt_weights_kernel(const float* __restrict__ W1, const float* __restrict__ W2,
                                   _Float16* __restrict__ W1t, _Float16* __restrict__ W2t) {
  int i = blockIdx.x * blockDim.x + threadIdx.x;
  const int n1 = IN_DIM * D1;          // 98304
  const int n2 = D1 * HID;             // 49152
  if (i < n1) {
    int k = i / D1, n = i % D1;
    W1t[(size_t)n * IN_DIM + k] = (_Float16)W1[i];
  } else if (i < n1 + n2) {
    int j = i - n1;
    int k = j / HID, n = j % HID;
    W2t[(size_t)n * D1 + k] = (_Float16)W2[j];
  }
}

// ---------------- WMMA GEMM:  C[M,N] = A[M,K] * Bt[N,K]^T  (f16 in, f32 out) ----------------
// one wave -> one 16x16 tile; per-lane A/B fragments per CDNA5 16-bit 16x32 layout:
//   lane l: row m = l&15, half = l>>4 ; element j (0..15): k = (j>>3)*16 + half*8 + (j&7)

template<int KDIM, int NDIM>
__global__ void gemm_wmma_kernel(const _Float16* __restrict__ A,
                                 const _Float16* __restrict__ Bt,
                                 float* __restrict__ C, int M) {
  const int lane = threadIdx.x & 31;
  const int wave = threadIdx.x >> 5;
  const int tilesN = NDIM / 16;
  const int mtiles = (M + 15) >> 4;
  const int tile = blockIdx.x * (blockDim.x >> 5) + wave;
  if (tile >= mtiles * tilesN) return;
  const int tm = tile / tilesN;
  const int tn = tile - tm * tilesN;

  const int r16  = lane & 15;
  const int half = lane >> 4;

  int arow = tm * 16 + r16;
  if (arow > M - 1) arow = M - 1;                       // clamp (store guarded below)
  const _Float16* __restrict__ Ap = A  + (size_t)arow * KDIM;
  const _Float16* __restrict__ Bp = Bt + (size_t)(tn * 16 + r16) * KDIM;

  v8f acc = {};
#pragma unroll
  for (int k0 = 0; k0 < KDIM; k0 += 32) {
    v8h alo = *(const v8h*)(Ap + k0 + half * 8);
    v8h ahi = *(const v8h*)(Ap + k0 + 16 + half * 8);
    v8h blo = *(const v8h*)(Bp + k0 + half * 8);
    v8h bhi = *(const v8h*)(Bp + k0 + 16 + half * 8);
    v16h a = __builtin_shufflevector(alo, ahi, 0,1,2,3,4,5,6,7,8,9,10,11,12,13,14,15);
    v16h b = __builtin_shufflevector(blo, bhi, 0,1,2,3,4,5,6,7,8,9,10,11,12,13,14,15);
    acc = __builtin_amdgcn_wmma_f32_16x16x32_f16(false, a, false, b,
                                                 (short)0, acc, false, false);
  }
  // C layout: VGPR r, lane l -> (m = tm*16 + r + 8*half, n = tn*16 + (l&15))
#pragma unroll
  for (int r = 0; r < 8; ++r) {
    int m = tm * 16 + r + 8 * half;
    if (m < M) C[(size_t)m * NDIM + tn * 16 + r16] = acc[r];
  }
}

// ---------------- attention projections ----------------

__global__ void proj_kernel(const float* __restrict__ h, const float* __restrict__ a_src,
                            const float* __restrict__ a_dst, float* __restrict__ s,
                            float* __restrict__ d, int Nn, int H) {
  int t = blockIdx.x * blockDim.x + threadIdx.x;
  if (t >= Nn * H) return;
  int n = t / H, hd = t - n * H;
  const float* __restrict__ hp = h + (size_t)n * H * HID + hd * HID;
  const float* __restrict__ as = a_src + hd * HID;
  const float* __restrict__ ad = a_dst + hd * HID;
  float accs = 0.f, accd = 0.f;
#pragma unroll 4
  for (int c = 0; c < HID; ++c) { float v = hp[c]; accs += v * as[c]; accd += v * ad[c]; }
  s[t] = accs; d[t] = accd;
}

// ---------------- segment softmax passes ----------------

template<int H>
__global__ void edge_max_kernel(const int* __restrict__ ei, int E, int Nn,
                                const float* __restrict__ s, const float* __restrict__ d,
                                float* __restrict__ elog, float* __restrict__ m) {
  int e = blockIdx.x * blockDim.x + threadIdx.x;
  if (e >= E + Nn) return;
  int sN, dN; get_edge(ei, E, Nn, e, sN, dN);
#pragma unroll
  for (int h = 0; h < H; ++h) {
    float v = leaky(s[sN * H + h] + d[dN * H + h]);
    elog[(size_t)e * H + h] = v;
    atomicMaxFloat(&m[dN * H + h], v);
  }
}

template<int H>
__global__ void edge_expsum_kernel(const int* __restrict__ ei, int E, int Nn,
                                   const float* __restrict__ elog, const float* __restrict__ m,
                                   float* __restrict__ den) {
  int e = blockIdx.x * blockDim.x + threadIdx.x;
  if (e >= E + Nn) return;
  int sN, dN; get_edge(ei, E, Nn, e, sN, dN);
#pragma unroll
  for (int h = 0; h < H; ++h)
    atomicAdd(&den[dN * H + h], expf(elog[(size_t)e * H + h] - m[dN * H + h]));
}

// conv1: alpha -> d_out section, plus node_attn = segsum(mean_h alpha, dst)
__global__ void edge_alpha1_kernel(const int* __restrict__ ei, int E, int Nn,
                                   const float* __restrict__ elog, const float* __restrict__ m,
                                   const float* __restrict__ den, float* __restrict__ alpha_out,
                                   float* __restrict__ natt) {
  int e = blockIdx.x * blockDim.x + threadIdx.x;
  if (e >= E + Nn) return;
  int sN, dN; get_edge(ei, E, Nn, e, sN, dN);
  float asum = 0.f;
#pragma unroll
  for (int h = 0; h < HEADS; ++h) {
    float a = expf(elog[(size_t)e * HEADS + h] - m[dN * HEADS + h]) / den[dN * HEADS + h];
    alpha_out[(size_t)e * HEADS + h] = a;
    asum += a;
  }
  atomicAdd(&natt[dN], asum * (1.0f / (float)HEADS));
}

// ---------------- aggregation ----------------

__global__ void agg1_kernel(const int* __restrict__ ei, int E, int Nn,
                            const float* __restrict__ h, const float* __restrict__ alpha,
                            float* __restrict__ out1) {
  int e = blockIdx.x;
  int c = threadIdx.x;                  // 128
  int sN, dN; get_edge(ei, E, Nn, e, sN, dN);
  float a0 = alpha[(size_t)e * 3 + 0];
  float a1 = alpha[(size_t)e * 3 + 1];
  float a2 = alpha[(size_t)e * 3 + 2];
  const float* __restrict__ hs = h + (size_t)sN * D1;
  float* __restrict__ od = out1 + (size_t)dN * D1;
  atomicAdd(&od[c],             hs[c]             * a0);
  atomicAdd(&od[HID + c],       hs[HID + c]       * a1);
  atomicAdd(&od[2 * HID + c],   hs[2 * HID + c]   * a2);
}

__global__ void agg2_kernel(const int* __restrict__ ei, int E, int Nn,
                            const float* __restrict__ h2, const float* __restrict__ elog,
                            const float* __restrict__ m2, const float* __restrict__ den2,
                            float* __restrict__ out2) {
  int e = blockIdx.x;
  int c = threadIdx.x;                  // 128
  int sN, dN; get_edge(ei, E, Nn, e, sN, dN);
  float a = expf(elog[e] - m2[dN]) / den2[dN];
  atomicAdd(&out2[(size_t)dN * HID + c], h2[(size_t)sN * HID + c] * a);
}

// bias + relu + convert to f16 (A matrix of GEMM2)
__global__ void bias_relu_cvt_kernel(const float* __restrict__ out1, const float* __restrict__ b1,
                                     _Float16* __restrict__ e1h, int n) {
  int i = blockIdx.x * blockDim.x + threadIdx.x;
  if (i >= n) return;
  float v = out1[i] + b1[i % D1];
  v = v > 0.f ? v : 0.f;
  e1h[i] = (_Float16)v;
}

// ---------------- pooling & similarity ----------------

// pooled[c]       = mean_n (e2[n,c] * natt[n])   (c < 128)
// pooled[128 + c] = mean_n  e2[n,c]
__global__ void pool_kernel(const float* __restrict__ out2, const float* __restrict__ b2,
                            const float* __restrict__ natt, float* __restrict__ pooled, int Nn) {
  int c = threadIdx.x;                  // 128
  int chunk = (Nn + gridDim.x - 1) / gridDim.x;
  int n0 = blockIdx.x * chunk;
  int n1 = n0 + chunk; if (n1 > Nn) n1 = Nn;
  float bw = b2[c], accw = 0.f, accm = 0.f;
  for (int n = n0; n < n1; ++n) {
    float v  = out2[(size_t)n * HID + c] + bw;
    float na = natt[n];
    accw += v * na;
    accm += v;
  }
  float inv = 1.0f / (float)Nn;
  atomicAdd(&pooled[c],       accw * inv);
  atomicAdd(&pooled[HID + c], accm * inv);
}

__global__ void sim_kernel(const float* __restrict__ pooled, float* __restrict__ out) {
  __shared__ float sdot[256], sna[256], snb[256];
  int t = threadIdx.x;                  // 256
  float a = pooled[t], b = pooled[256 + t];
  sdot[t] = a * b; sna[t] = a * a; snb[t] = b * b;
  __syncthreads();
  for (int s = 128; s > 0; s >>= 1) {
    if (t < s) { sdot[t] += sdot[t + s]; sna[t] += sna[t + s]; snb[t] += snb[t + s]; }
    __syncthreads();
  }
  if (t == 0) {
    float n1 = fmaxf(sqrtf(sna[0]), 1e-8f);
    float n2 = fmaxf(sqrtf(snb[0]), 1e-8f);
    out[0] = sdot[0] / (n1 * n2);
  }
}

// ---------------------------------------------------------------------------

extern "C" void kernel_launch(void* const* d_in, const int* in_sizes, int n_in,
                              void* d_out, int out_size, void* d_ws, size_t ws_size,
                              hipStream_t stream) {
  const float* x[2]  = { (const float*)d_in[0], (const float*)d_in[2] };
  const int*   ei[2] = { (const int*)d_in[1],   (const int*)d_in[3]   };
  const float* W1  = (const float*)d_in[4];
  const float* as1 = (const float*)d_in[5];
  const float* ad1 = (const float*)d_in[6];
  const float* b1  = (const float*)d_in[7];
  const float* W2  = (const float*)d_in[8];
  const float* as2 = (const float*)d_in[9];
  const float* ad2 = (const float*)d_in[10];
  const float* b2  = (const float*)d_in[11];

  const int Nn   = in_sizes[0] / IN_DIM;   // 10000
  const int E    = in_sizes[1] / 2;        // 320000
  const int Etot = E + Nn;                 // 330000

  // ---- workspace layout (256B aligned) ----
  char* ws = (char*)d_ws;
  size_t off = 0;
  auto alloc = [&](size_t bytes) { size_t o = off; off += (bytes + 255) & ~(size_t)255; return o; };
  _Float16* W1t  = (_Float16*)(ws + alloc((size_t)D1 * IN_DIM * 2));
  _Float16* W2t  = (_Float16*)(ws + alloc((size_t)HID * D1 * 2));
  _Float16* xh   = (_Float16*)(ws + alloc((size_t)Nn * IN_DIM * 2));
  float*    hbuf = (float*)   (ws + alloc((size_t)Nn * D1 * 4));
  float*    sbuf = (float*)   (ws + alloc((size_t)Nn * HEADS * 4));
  float*    dbuf = (float*)   (ws + alloc((size_t)Nn * HEADS * 4));
  float*    elog = (float*)   (ws + alloc((size_t)Etot * HEADS * 4));
  float*    mbuf = (float*)   (ws + alloc((size_t)Nn * HEADS * 4));
  float*    den  = (float*)   (ws + alloc((size_t)Nn * HEADS * 4));
  float*    out1 = (float*)   (ws + alloc((size_t)Nn * D1 * 4));
  _Float16* e1h  = (_Float16*)(ws + alloc((size_t)Nn * D1 * 2));
  float*    natt = (float*)   (ws + alloc((size_t)Nn * 4));
  float*    h2   = (float*)   (ws + alloc((size_t)Nn * HID * 4));
  float*    s2   = (float*)   (ws + alloc((size_t)Nn * 4));
  float*    d2   = (float*)   (ws + alloc((size_t)Nn * 4));
  float*    e2l  = (float*)   (ws + alloc((size_t)Etot * 4));
  float*    m2   = (float*)   (ws + alloc((size_t)Nn * 4));
  float*    den2 = (float*)   (ws + alloc((size_t)Nn * 4));
  float*    out2 = (float*)   (ws + alloc((size_t)Nn * HID * 4));
  float*    pooled = (float*) (ws + alloc(2 * 256 * 4));

  float* outp = (float*)d_out;

  const int TB = 256;
  const int egrid = (Etot + TB - 1) / TB;

  // ---- shared prep ----
  cvt_weights_kernel<<<(IN_DIM * D1 + D1 * HID + TB - 1) / TB, TB, 0, stream>>>(W1, W2, W1t, W2t);
  hipMemsetAsync(pooled, 0, 2 * 256 * sizeof(float), stream);

  for (int b = 0; b < 2; ++b) {
    float* alpha_out = outp + 1 + (size_t)b * Etot * HEADS;

    // init accumulators
    hipMemsetAsync(den,  0, (size_t)Nn * HEADS * 4, stream);
    hipMemsetAsync(out1, 0, (size_t)Nn * D1 * 4, stream);
    hipMemsetAsync(natt, 0, (size_t)Nn * 4, stream);
    hipMemsetAsync(den2, 0, (size_t)Nn * 4, stream);
    hipMemsetAsync(out2, 0, (size_t)Nn * HID * 4, stream);
    fill_f32_kernel<<<64, TB, 0, stream>>>(mbuf, Nn * HEADS, -INFINITY);
    fill_f32_kernel<<<64, TB, 0, stream>>>(m2, Nn, -INFINITY);

    // ---- GATConv 1 ----
    cvt_x_kernel<<<(Nn * IN_DIM + TB - 1) / TB, TB, 0, stream>>>(x[b], xh, Nn * IN_DIM);
    {
      int tiles = ((Nn + 15) / 16) * (D1 / 16);
      gemm_wmma_kernel<IN_DIM, D1><<<(tiles + 7) / 8, TB, 0, stream>>>(xh, W1t, hbuf, Nn);
    }
    proj_kernel<<<(Nn * HEADS + TB - 1) / TB, TB, 0, stream>>>(hbuf, as1, ad1, sbuf, dbuf, Nn, HEADS);
    edge_max_kernel<HEADS><<<egrid, TB, 0, stream>>>(ei[b], E, Nn, sbuf, dbuf, elog, mbuf);
    edge_expsum_kernel<HEADS><<<egrid, TB, 0, stream>>>(ei[b], E, Nn, elog, mbuf, den);
    edge_alpha1_kernel<<<egrid, TB, 0, stream>>>(ei[b], E, Nn, elog, mbuf, den, alpha_out, natt);
    agg1_kernel<<<Etot, HID, 0, stream>>>(ei[b], E, Nn, hbuf, alpha_out, out1);
    bias_relu_cvt_kernel<<<(Nn * D1 + TB - 1) / TB, TB, 0, stream>>>(out1, b1, e1h, Nn * D1);

    // ---- GATConv 2 ----
    {
      int tiles = ((Nn + 15) / 16) * (HID / 16);
      gemm_wmma_kernel<D1, HID><<<(tiles + 7) / 8, TB, 0, stream>>>(e1h, W2t, h2, Nn);
    }
    proj_kernel<<<(Nn + TB - 1) / TB, TB, 0, stream>>>(h2, as2, ad2, s2, d2, Nn, 1);
    edge_max_kernel<1><<<egrid, TB, 0, stream>>>(ei[b], E, Nn, s2, d2, e2l, m2);
    edge_expsum_kernel<1><<<egrid, TB, 0, stream>>>(ei[b], E, Nn, e2l, m2, den2);
    agg2_kernel<<<Etot, HID, 0, stream>>>(ei[b], E, Nn, h2, e2l, m2, den2, out2);

    // ---- pooling ----
    pool_kernel<<<80, HID, 0, stream>>>(out2, b2, natt, pooled + (size_t)b * 256, Nn);
  }

  sim_kernel<<<1, 256, 0, stream>>>(pooled, outp);
}
